// PIDProjectionV3_627065225877
// MI455X (gfx1250) — matrix operand order repaired and checked by reading, not compiled, not run on hardware
//
#include <hip/hip_runtime.h>
#include <hip/hip_bf16.h>
#include <math.h>

typedef __attribute__((ext_vector_type(16))) __bf16 v16bf;
typedef __attribute__((ext_vector_type(8)))  float  v8f;
typedef __attribute__((ext_vector_type(4)))  int    v4i_t;
typedef unsigned short ushort_t;
typedef unsigned int   uint_t;

#define AS_GLOBAL __attribute__((address_space(1)))
#define AS_LDS    __attribute__((address_space(3)))

#define B_DIM 4
#define T_DIM 2048
#define D_DIM 2048
#define K_DIM 6144          // 3 * D_DIM
#define M_DIM 8192          // B * T
#define I_DECAY 0.95f
#define STAG_THRESH 0.95f
#define MAX_GATE 0.6f
#define MIN_I_GATE 0.25f
#define EPSV 1e-12f

#if defined(__has_builtin)
#if __has_builtin(__builtin_amdgcn_global_load_async_to_lds_b128) && \
    __has_builtin(__builtin_amdgcn_s_wait_asynccnt)
#define USE_ASYNC_LDS 1
#endif
#endif
#ifndef USE_ASYNC_LDS
#define USE_ASYNC_LDS 0
#endif

__device__ __forceinline__ ushort_t f2bf(float f) {
    uint_t u = __builtin_bit_cast(uint_t, f);
    u += 0x7FFFu + ((u >> 16) & 1u);   // round-to-nearest-even
    return (ushort_t)(u >> 16);
}

// ---------------------------------------------------------------------------
// Kernel 1: per-(b,t) scalars: gates (softmax->clamp->renorm) and stag*kick
// ---------------------------------------------------------------------------
__global__ void k_scalars(const float* __restrict__ x,
                          const float* __restrict__ gate_w,
                          const float* __restrict__ gate_b,
                          const float* __restrict__ kick_w,
                          const float* __restrict__ kick_b,
                          float4* __restrict__ d_s) {
    const int t = blockIdx.x;
    const int b = blockIdx.y;
    const int tid = threadIdx.x;
    const float* xt  = x + ((size_t)b * T_DIM + t) * D_DIM;
    const float* xtm = (t > 0) ? (xt - D_DIM) : xt;

    float a0 = 0.f, a1 = 0.f, a2 = 0.f, ak = 0.f;
    float ap = 0.f, ant = 0.f, anp = 0.f;
    for (int i = tid; i < D_DIM; i += 256) {
        float xv = xt[i];
        float xp = xtm[i];
        a0  += xv * gate_w[i * 3 + 0];
        a1  += xv * gate_w[i * 3 + 1];
        a2  += xv * gate_w[i * 3 + 2];
        ak  += xv * kick_w[i];
        ap  += xv * xp;
        ant += xv * xv;
        anp += xp * xp;
    }

    __shared__ float4 red[256];
    red[tid] = make_float4(a0, a1, a2, ak);
    __syncthreads();
    for (int s = 128; s > 0; s >>= 1) {
        if (tid < s) {
            float4 o = red[tid + s];
            red[tid].x += o.x; red[tid].y += o.y;
            red[tid].z += o.z; red[tid].w += o.w;
        }
        __syncthreads();
    }
    float4 r0 = red[0];
    __syncthreads();
    red[tid] = make_float4(ap, ant, anp, 0.f);
    __syncthreads();
    for (int s = 128; s > 0; s >>= 1) {
        if (tid < s) {
            float4 o = red[tid + s];
            red[tid].x += o.x; red[tid].y += o.y;
            red[tid].z += o.z; red[tid].w += o.w;
        }
        __syncthreads();
    }

    if (tid == 0) {
        float4 r1 = red[0];
        float l0 = r0.x + gate_b[0], l1 = r0.y + gate_b[1], l2 = r0.z + gate_b[2];
        float mx = fmaxf(l0, fmaxf(l1, l2));
        float e0 = expf(l0 - mx), e1 = expf(l1 - mx), e2 = expf(l2 - mx);
        float inv = 1.f / (e0 + e1 + e2);
        float g0 = e0 * inv, g1 = e1 * inv, g2 = e2 * inv;
        g0 = fminf(g0, MAX_GATE); g1 = fminf(g1, MAX_GATE); g2 = fminf(g2, MAX_GATE);
        g1 = fmaxf(g1, MIN_I_GATE);
        float gs = 1.f / (g0 + g1 + g2);
        g0 *= gs; g1 *= gs; g2 *= gs;
        float kick = 1.f / (1.f + expf(-(r0.w + kick_b[0])));
        float s3 = 0.f;
        if (t > 0) {
            float nt = fmaxf(sqrtf(r1.y), EPSV);
            float np = fmaxf(sqrtf(r1.z), EPSV);
            float cs = r1.x / (nt * np);
            s3 = (cs > STAG_THRESH) ? kick : 0.f;
        }
        d_s[(size_t)b * T_DIM + t] = make_float4(g0, g1, g2, s3);
    }
}

// ---------------------------------------------------------------------------
// Kernel 2: build Wt[n][k] (bf16, K-major) = transpose of stacked [Wp;Wi;Wd]
// ---------------------------------------------------------------------------
__global__ void k_wt(const float* __restrict__ Wp,
                     const float* __restrict__ Wi,
                     const float* __restrict__ Wd,
                     ushort_t* __restrict__ d_Wt) {
    __shared__ float tile[32][33];
    const int kT = blockIdx.x * 32;
    const int nT = blockIdx.y * 32;
    const int tx = threadIdx.x & 31;
    const int ty0 = threadIdx.x >> 5;

    for (int j = 0; j < 4; ++j) {
        int ty = ty0 + j * 8;
        int k = kT + ty;
        const float* W = (k < D_DIM) ? (Wp + (size_t)k * D_DIM)
                       : (k < 2 * D_DIM) ? (Wi + (size_t)(k - D_DIM) * D_DIM)
                       : (Wd + (size_t)(k - 2 * D_DIM) * D_DIM);
        tile[ty][tx] = W[nT + tx];
    }
    __syncthreads();
    for (int j = 0; j < 4; ++j) {
        int ty = ty0 + j * 8;
        d_Wt[(size_t)(nT + ty) * K_DIM + (kT + tx)] = f2bf(tile[tx][ty]);
    }
}

// ---------------------------------------------------------------------------
// Kernel 3: EMA scan over T per (b,d) channel; emit bf16 A = [g0*x|g1*I|g2*D]
// ---------------------------------------------------------------------------
__global__ void k_scan(const float* __restrict__ x,
                       const float* __restrict__ kick_vec,
                       const float4* __restrict__ d_s,
                       ushort_t* __restrict__ d_A) {
    const int ch = blockIdx.x * 256 + threadIdx.x;
    const int b = ch >> 11;
    const int d = ch & (D_DIM - 1);
    const float kv = kick_vec[d];
    const float* xc = x + (size_t)b * T_DIM * D_DIM + d;
    const float4* sc = d_s + (size_t)b * T_DIM;

    float integ = 0.f, xprev = 0.f;
    for (int t = 0; t < T_DIM; ++t) {
        __builtin_prefetch(&xc[(size_t)(t + 8) * D_DIM], 0, 0);
        float xv = xc[(size_t)t * D_DIM];
        float4 s = sc[t];
        integ = (t == 0) ? xv : (I_DECAY * integ + (1.f - I_DECAY) * xv);
        float diff  = (t == 0) ? 0.f : (xv - xprev);
        float deriv = diff + s.w * kv;
        ushort_t* Arow = d_A + (size_t)((size_t)b * T_DIM + t) * K_DIM;
        Arow[d]             = f2bf(s.x * xv);
        Arow[D_DIM + d]     = f2bf(s.y * integ);
        Arow[2 * D_DIM + d] = f2bf(s.z * deriv);
        xprev = xv;
    }
}

// ---------------------------------------------------------------------------
// Kernel 4: bf16 WMMA GEMM: out[8192][2048] = A[M][6144] * Wt[N][6144]^T + bias
// 256 threads (8 waves), 128x128 block tile, waves 4(M)x2(N), wave tile 32x64.
// LDS rows padded to 40 halves (80B = 5*16B -> aligned b128, bank-rotated).
// Staging uses GLOBAL_LOAD_ASYNC_TO_LDS_B128 (ASYNCcnt) when available.
// ---------------------------------------------------------------------------
#define LDP 40

__device__ __forceinline__ v16bf ld_frag(const ushort_t* p0, const ushort_t* p1) {
    struct P { uint4 a, b; } s;
    s.a = *(const uint4*)p0;
    s.b = *(const uint4*)p1;
    return __builtin_bit_cast(v16bf, s);
}

__global__ void __launch_bounds__(256)
k_gemm(const ushort_t* __restrict__ d_A,
       const ushort_t* __restrict__ d_Wt,
       const float* __restrict__ bias,
       float* __restrict__ out) {
    __shared__ __align__(16) ushort_t lsA[2][128 * LDP];
    __shared__ __align__(16) ushort_t lsB[2][128 * LDP];

    const int tid  = threadIdx.x;
    const int lane = tid & 31;
    const int wave = tid >> 5;
    const int waveM = wave >> 1;   // 0..3
    const int waveN = wave & 1;    // 0..1
    const int nBase = blockIdx.x * 128;
    const int mBase = blockIdx.y * 128;

    v8f acc[2][4];
    for (int i = 0; i < 2; ++i)
        for (int j = 0; j < 4; ++j)
            acc[i][j] = {};

    auto stage = [&](int buf, int k0) {
        for (int i = 0; i < 2; ++i) {
            int chunk = tid + i * 256;          // 0..511
            int row  = chunk >> 2;              // 0..127
            int colh = (chunk & 3) * 8;         // half-index, 16B aligned
            const ushort_t* gA = &d_A [(size_t)(mBase + row) * K_DIM + k0 + colh];
            const ushort_t* gB = &d_Wt[(size_t)(nBase + row) * K_DIM + k0 + colh];
            ushort_t* lA = &lsA[buf][row * LDP + colh];
            ushort_t* lB = &lsB[buf][row * LDP + colh];
#if USE_ASYNC_LDS
            __builtin_amdgcn_global_load_async_to_lds_b128(
                (AS_GLOBAL v4i_t*)gA, (AS_LDS v4i_t*)lA, 0, 0);
            __builtin_amdgcn_global_load_async_to_lds_b128(
                (AS_GLOBAL v4i_t*)gB, (AS_LDS v4i_t*)lB, 0, 0);
#else
            *(uint4*)lA = *(const uint4*)gA;
            *(uint4*)lB = *(const uint4*)gB;
#endif
        }
    };

    auto stage_fence = [&]() {
#if USE_ASYNC_LDS
        __builtin_amdgcn_s_wait_asynccnt(0);
#endif
        __syncthreads();
    };

    stage(0, 0);
    stage_fence();

    const int r  = lane & 15;
    const int hs = lane >> 4;
    const int NK = K_DIM / 32;    // 192 k-steps

    for (int kt = 0; kt < NK; ++kt) {
        int buf = kt & 1;
        if (kt + 1 < NK) stage(buf ^ 1, (kt + 1) * 32);

        const ushort_t* As = lsA[buf];
        const ushort_t* Bs = lsB[buf];

        // A frag: lanes 0-15 row r K{0-7,16-23}; lanes 16-31 row r K{8-15,24-31}
        v16bf aF[2];
        for (int tm = 0; tm < 2; ++tm) {
            int rowA = (waveM * 32 + tm * 16 + r) * LDP;
            aF[tm] = ld_frag(&As[rowA + hs * 8], &As[rowA + 16 + hs * 8]);
        }
        // B frag: lane col n=r; lanes 0-15 K{0-7,16-23}, lanes 16-31 K{8-15,24-31} (K-major LDS)
        v16bf bF[4];
        for (int tn = 0; tn < 4; ++tn) {
            int rowB = (waveN * 64 + tn * 16 + r) * LDP;
            bF[tn] = ld_frag(&Bs[rowB + hs * 8], &Bs[rowB + 16 + hs * 8]);
        }

        for (int tm = 0; tm < 2; ++tm)
            for (int tn = 0; tn < 4; ++tn)
                acc[tm][tn] = __builtin_amdgcn_wmma_f32_16x16x32_bf16(
                    false, aF[tm], false, bF[tn], (short)0, acc[tm][tn],
                    false, false);

        stage_fence();
    }

    // Epilogue: C/D layout -> VGPR e: lanes0-15 M=e, lanes16-31 M=e+8; N=lane&15
    for (int tm = 0; tm < 2; ++tm) {
        for (int tn = 0; tn < 4; ++tn) {
            int n = nBase + waveN * 64 + tn * 16 + r;
            float bn = bias[n];
            for (int e = 0; e < 8; ++e) {
                int m = mBase + waveM * 32 + tm * 16 + e + 8 * hs;
                out[(size_t)m * D_DIM + n] = acc[tm][tn][e] + bn;
            }
        }
    }
}

// ---------------------------------------------------------------------------
extern "C" void kernel_launch(void* const* d_in, const int* in_sizes, int n_in,
                              void* d_out, int out_size, void* d_ws, size_t ws_size,
                              hipStream_t stream) {
    const float* x        = (const float*)d_in[0];
    const float* Wp       = (const float*)d_in[1];
    const float* Wi       = (const float*)d_in[2];
    const float* Wd       = (const float*)d_in[3];
    const float* gate_w   = (const float*)d_in[4];
    const float* gate_b   = (const float*)d_in[5];
    const float* kick_vec = (const float*)d_in[6];
    const float* kick_w   = (const float*)d_in[7];
    const float* kick_b   = (const float*)d_in[8];
    const float* bias     = (const float*)d_in[9];
    float* out = (float*)d_out;

    char* ws = (char*)d_ws;
    float4*   d_s  = (float4*)ws;                                        // 128KB
    ushort_t* d_Wt = (ushort_t*)(ws + 131072);                           // 24MB
    ushort_t* d_A  = (ushort_t*)(ws + 131072 + (size_t)K_DIM * D_DIM * 2); // 96MB

    k_scalars<<<dim3(T_DIM, B_DIM), 256, 0, stream>>>(x, gate_w, gate_b,
                                                      kick_w, kick_b, d_s);
    k_wt<<<dim3(K_DIM / 32, D_DIM / 32), 256, 0, stream>>>(Wp, Wi, Wd, d_Wt);
    k_scan<<<dim3(M_DIM / 256), 256, 0, stream>>>(x, kick_vec, d_s, d_A);
    k_gemm<<<dim3(D_DIM / 128, M_DIM / 128), 256, 0, stream>>>(d_A, d_Wt, bias, out);
}
